// FirFilterNet_11570641896040
// MI455X (gfx1250) — matrix operand order repaired and checked by reading, not compiled, not run on hardware
//
#include <hip/hip_runtime.h>
#include <hip/hip_bf16.h>

// ---------------- static problem sizes ----------------
#define N_WINDOW   1024
#define FRAME_SIZE 960
#define C_CONTENT  192
#define C_SPK      256
#define C_INT      512
#define KSIZE      7
#define N_LAYERS   6
#define D_MLP      (C_INT * 3)      // 1536
#define BATCH      8
#define N_FRAMES   512
#define BT         (BATCH * N_FRAMES)  // 4096 rows
#define LLEN       (N_FRAMES * FRAME_SIZE) // 491520

typedef __attribute__((ext_vector_type(16))) __bf16 v16bf;
typedef __attribute__((ext_vector_type(8)))  float  v8f;

union BF16x16 { uint4 u[2]; v16bf v; };

__device__ __forceinline__ unsigned short f2bf(float f) {
    unsigned int u = __float_as_uint(f);
    u += 0x7fffu + ((u >> 16) & 1u);     // round-to-nearest-even
    return (unsigned short)(u >> 16);
}

__device__ __forceinline__ float gelu_exact(float x) {
    return 0.5f * x * (1.0f + erff(x * 0.70710678118654752f));
}

// ---------------- weight conversion ----------------
__global__ void f2bf_copy_kernel(const float* __restrict__ in,
                                 unsigned short* __restrict__ out, int n) {
    int i = blockIdx.x * 256 + threadIdx.x;
    if (i < n) out[i] = f2bf(in[i]);
}

// w_out: [1024][512][7] fp32  ->  [7][1024][512] bf16
__global__ void wout_tr_kernel(const float* __restrict__ w,
                               unsigned short* __restrict__ o) {
    int i = blockIdx.x * 256 + threadIdx.x;
    const int total = N_WINDOW * C_INT * KSIZE;
    if (i < total) {
        int k  = i % KSIZE;
        int c  = (i / KSIZE) % C_INT;
        int wo = i / (KSIZE * C_INT);
        o[(size_t)k * N_WINDOW * C_INT + (size_t)wo * C_INT + c] = f2bf(w[i]);
    }
}

// ---------------- conditioning trunk: x[bt][o] fp32 ----------------
__global__ void trunk_kernel(const float* __restrict__ content,
                             const float* __restrict__ f0,
                             const float* __restrict__ energy,
                             const float* __restrict__ spk,
                             const float* __restrict__ w_content,
                             const float* __restrict__ b_content,
                             const float* __restrict__ w_spk,
                             const float* __restrict__ b_spk,
                             const float* __restrict__ w_f0,
                             const float* __restrict__ b_f0,
                             const float* __restrict__ w_en,
                             const float* __restrict__ b_en,
                             float* __restrict__ xf) {
    int bt = blockIdx.x;
    int b = bt >> 9, t = bt & 511;
    int o = blockIdx.y * 256 + threadIdx.x;          // gridDim.y == 2
    float f0v = f0[b * N_FRAMES + t];
    float lf0 = logf(fmaxf(f0v, 0.0f) + 1e-6f);
    float env = energy[b * N_FRAMES + t];
    float acc = b_content[o] + b_spk[o] + w_f0[o] * lf0 + b_f0[o]
              + w_en[o] * env + b_en[o];
    const float* cb = content + (size_t)b * C_CONTENT * N_FRAMES + t;
    const float* wc = w_content + (size_t)o * C_CONTENT;
    #pragma unroll 4
    for (int c = 0; c < C_CONTENT; ++c) acc += wc[c] * cb[(size_t)c * N_FRAMES];
    const float* sb = spk + (size_t)b * C_SPK;
    const float* ws = w_spk + (size_t)o * C_SPK;
    #pragma unroll 4
    for (int c = 0; c < C_SPK; ++c) acc += ws[c] * sb[c];
    xf[(size_t)bt * C_INT + o] = acc;
}

// ---------------- fused depthwise conv k=7 (zero pad) + channel-LN -> bf16 ----------------
__global__ __launch_bounds__(256)
void dwln_kernel(const float* __restrict__ xf, const float* __restrict__ dw_w,
                 const float* __restrict__ dw_b, const float* __restrict__ g,
                 const float* __restrict__ bb, unsigned short* __restrict__ h_bf) {
    int bt = blockIdx.x;
    int b = bt >> 9, t = bt & 511;
    __shared__ float sh[C_INT];
    __shared__ float s1[256], s2[256];
    int tid = threadIdx.x;
    float ls = 0.f, lq = 0.f;
    for (int c = tid; c < C_INT; c += 256) {
        float acc = dw_b[c];
        #pragma unroll
        for (int k = 0; k < KSIZE; ++k) {
            int tt = t + k - (KSIZE / 2);
            if (tt >= 0 && tt < N_FRAMES)
                acc += dw_w[c * KSIZE + k] * xf[((size_t)(b * N_FRAMES + tt)) * C_INT + c];
        }
        sh[c] = acc;
        ls += acc; lq += acc * acc;
    }
    s1[tid] = ls; s2[tid] = lq;
    __syncthreads();
    for (int off = 128; off >= 1; off >>= 1) {
        if (tid < off) { s1[tid] += s1[tid + off]; s2[tid] += s2[tid + off]; }
        __syncthreads();
    }
    float mu = s1[0] * (1.0f / C_INT);
    float var = s2[0] * (1.0f / C_INT) - mu * mu;
    float rs = rsqrtf(var + 1e-6f);
    for (int c = tid; c < C_INT; c += 256) {
        float v = (sh[c] - mu) * rs * g[c] + bb[c];
        h_bf[(size_t)bt * C_INT + c] = f2bf(v);
    }
}

// ---------------- row LayerNorm -> bf16 (output norm) ----------------
__global__ __launch_bounds__(256)
void rowln_kernel(const float* __restrict__ xf, const float* __restrict__ g,
                  const float* __restrict__ bb, unsigned short* __restrict__ out_bf) {
    int bt = blockIdx.x;
    __shared__ float s1[256], s2[256];
    int tid = threadIdx.x;
    float ls = 0.f, lq = 0.f;
    for (int c = tid; c < C_INT; c += 256) {
        float v = xf[(size_t)bt * C_INT + c];
        ls += v; lq += v * v;
    }
    s1[tid] = ls; s2[tid] = lq;
    __syncthreads();
    for (int off = 128; off >= 1; off >>= 1) {
        if (tid < off) { s1[tid] += s1[tid + off]; s2[tid] += s2[tid + off]; }
        __syncthreads();
    }
    float mu = s1[0] * (1.0f / C_INT);
    float var = s2[0] * (1.0f / C_INT) - mu * mu;
    float rs = rsqrtf(var + 1e-6f);
    for (int c = tid; c < C_INT; c += 256) {
        float v = (xf[(size_t)bt * C_INT + c] - mu) * rs * g[c] + bb[c];
        out_bf[(size_t)bt * C_INT + c] = f2bf(v);
    }
}

// ---------------- WMMA bf16 GEMM:  Out[N][M] = A[M][K] * Bt[N][K]^T + bias ----------------
// mode 0: + GELU, store bf16 [N][M]        (pw1)
// mode 1: + residual (Res[N][M] f32), store f32 [N][M]   (pw2)
__global__ __launch_bounds__(256)
void gemm_bf16_kernel(const unsigned short* __restrict__ A,
                      const unsigned short* __restrict__ Bt,
                      const float* __restrict__ bias,
                      void* __restrict__ Out,
                      const float* __restrict__ Res,
                      int M, int K, int mode) {
    int lane = threadIdx.x & 31;
    int wave = threadIdx.x >> 5;
    int wm = wave & 3, wn = wave >> 2;
    int m_base = blockIdx.x * 128 + wm * 32;
    int n_base = blockIdx.y * 64 + wn * 32;
    int lh = lane & 15, sel = lane >> 4;

    v8f acc[2][2];
    #pragma unroll
    for (int i = 0; i < 2; ++i)
        #pragma unroll
        for (int j = 0; j < 2; ++j)
            #pragma unroll
            for (int r = 0; r < 8; ++r) acc[i][j][r] = 0.0f;

    for (int k0 = 0; k0 < K; k0 += 32) {
        BF16x16 a[2], b[2];
        #pragma unroll
        for (int i = 0; i < 2; ++i) {
            const uint4* p = (const uint4*)(A + (size_t)(m_base + i * 16 + lh) * K + k0 + sel * 8);
            a[i].u[0] = p[0];
            a[i].u[1] = p[2];     // +16 bf16 elements
        }
        #pragma unroll
        for (int j = 0; j < 2; ++j) {
            const uint4* p = (const uint4*)(Bt + (size_t)(n_base + j * 16 + lh) * K + k0 + sel * 8);
            b[j].u[0] = p[0];
            b[j].u[1] = p[2];
        }
        #pragma unroll
        for (int i = 0; i < 2; ++i)
            #pragma unroll
            for (int j = 0; j < 2; ++j)
                acc[i][j] = __builtin_amdgcn_wmma_f32_16x16x32_bf16(
                    false, a[i].v, false, b[j].v, (short)0, acc[i][j], false, false);
    }

    #pragma unroll
    for (int i = 0; i < 2; ++i) {
        #pragma unroll
        for (int j = 0; j < 2; ++j) {
            int n  = n_base + j * 16 + lh;
            int m0 = m_base + i * 16 + sel * 8;
            float v[8];
            #pragma unroll
            for (int r = 0; r < 8; ++r) v[r] = acc[i][j][r] + bias[m0 + r];
            if (mode == 0) {
                #pragma unroll
                for (int r = 0; r < 8; ++r) v[r] = gelu_exact(v[r]);
                uint4 st;
                st.x = (unsigned)f2bf(v[0]) | ((unsigned)f2bf(v[1]) << 16);
                st.y = (unsigned)f2bf(v[2]) | ((unsigned)f2bf(v[3]) << 16);
                st.z = (unsigned)f2bf(v[4]) | ((unsigned)f2bf(v[5]) << 16);
                st.w = (unsigned)f2bf(v[6]) | ((unsigned)f2bf(v[7]) << 16);
                *(uint4*)((unsigned short*)Out + (size_t)n * M + m0) = st;
            } else {
                const float4* rp = (const float4*)(Res + (size_t)n * M + m0);
                float4 r0 = rp[0], r1 = rp[1];
                float4 o0, o1;
                o0.x = v[0] + r0.x; o0.y = v[1] + r0.y; o0.z = v[2] + r0.z; o0.w = v[3] + r0.w;
                o1.x = v[4] + r1.x; o1.y = v[5] + r1.y; o1.z = v[6] + r1.z; o1.w = v[7] + r1.w;
                float4* op = (float4*)((float*)Out + (size_t)n * M + m0);
                op[0] = o0; op[1] = o1;
            }
        }
    }
}

// ---------------- filter head: 7 shifted GEMMs (replicate pad), f32 out ----------------
// A7: [7][1024][512] bf16, Bt: [4096][512] bf16, filt: [4096][1024] f32
__global__ __launch_bounds__(256)
void gemm_filt_kernel(const unsigned short* __restrict__ A7,
                      const unsigned short* __restrict__ Bt,
                      const float* __restrict__ bias,
                      float* __restrict__ filt) {
    const int M = N_WINDOW, K = C_INT;
    int lane = threadIdx.x & 31;
    int wave = threadIdx.x >> 5;
    int wm = wave & 3, wn = wave >> 2;
    int m_base = blockIdx.x * 128 + wm * 32;
    int n_base = blockIdx.y * 64 + wn * 32;
    int lh = lane & 15, sel = lane >> 4;

    v8f acc[2][2];
    #pragma unroll
    for (int i = 0; i < 2; ++i)
        #pragma unroll
        for (int j = 0; j < 2; ++j)
            #pragma unroll
            for (int r = 0; r < 8; ++r) acc[i][j][r] = 0.0f;

    for (int kk = 0; kk < KSIZE; ++kk) {
        const unsigned short* A = A7 + (size_t)kk * M * K;
        // per-lane remapped B rows (replicate padding within each batch of 512 frames)
        int rrow[2];
        #pragma unroll
        for (int j = 0; j < 2; ++j) {
            int n = n_base + j * 16 + lh;
            int b = n >> 9, t = n & 511;
            int tt = t + kk - (KSIZE / 2);
            tt = tt < 0 ? 0 : (tt > 511 ? 511 : tt);
            rrow[j] = (b << 9) | tt;
        }
        for (int k0 = 0; k0 < K; k0 += 32) {
            BF16x16 a[2], b[2];
            #pragma unroll
            for (int i = 0; i < 2; ++i) {
                const uint4* p = (const uint4*)(A + (size_t)(m_base + i * 16 + lh) * K + k0 + sel * 8);
                a[i].u[0] = p[0];
                a[i].u[1] = p[2];
            }
            #pragma unroll
            for (int j = 0; j < 2; ++j) {
                const uint4* p = (const uint4*)(Bt + (size_t)rrow[j] * K + k0 + sel * 8);
                b[j].u[0] = p[0];
                b[j].u[1] = p[2];
            }
            #pragma unroll
            for (int i = 0; i < 2; ++i)
                #pragma unroll
                for (int j = 0; j < 2; ++j)
                    acc[i][j] = __builtin_amdgcn_wmma_f32_16x16x32_bf16(
                        false, a[i].v, false, b[j].v, (short)0, acc[i][j], false, false);
        }
    }

    #pragma unroll
    for (int i = 0; i < 2; ++i) {
        #pragma unroll
        for (int j = 0; j < 2; ++j) {
            int n  = n_base + j * 16 + lh;
            int m0 = m_base + i * 16 + sel * 8;
            float4 o0, o1;
            o0.x = acc[i][j][0] + bias[m0 + 0];
            o0.y = acc[i][j][1] + bias[m0 + 1];
            o0.z = acc[i][j][2] + bias[m0 + 2];
            o0.w = acc[i][j][3] + bias[m0 + 3];
            o1.x = acc[i][j][4] + bias[m0 + 4];
            o1.y = acc[i][j][5] + bias[m0 + 5];
            o1.z = acc[i][j][6] + bias[m0 + 6];
            o1.w = acc[i][j][7] + bias[m0 + 7];
            float4* op = (float4*)(filt + (size_t)n * M + m0);
            op[0] = o0; op[1] = o1;
        }
    }
}

// ---------------- per-frame FIR + overlap-add (gather form, no atomics) ----------------
// out[b][p] = sum_{q=p+1}^{min(p+1024, L-1)} src[b][q] * filt[b][q/960][1024 + p - q]
// Register-blocked: 4 consecutive outputs per thread share each src sample (4 FMAs / src),
// src window staged in LDS, 4 consecutive taps per src sample (same cacheline, lane-invariant).
#define FIR_NP     4
#define FIR_BLOCK  256
#define FIR_PTILE  (FIR_BLOCK * FIR_NP)   // 1024 outputs per block
#define FIR_QWIN   (FIR_PTILE + N_WINDOW) // 2048-float source window

__global__ __launch_bounds__(FIR_BLOCK)
void fir_kernel(const float* __restrict__ src, const float* __restrict__ filt,
                float* __restrict__ out) {
    __shared__ float sbuf[FIR_QWIN];
    int b  = blockIdx.y;
    int P0 = blockIdx.x * FIR_PTILE;
    const float* s  = src  + (size_t)b * LLEN;
    const float* fb = filt + (size_t)b * N_FRAMES * N_WINDOW;

    // stage source window covering q in [P0+1, P0+FIR_QWIN]
    for (int d = threadIdx.x; d < FIR_QWIN; d += FIR_BLOCK) {
        int q = P0 + 1 + d;
        sbuf[d] = (q < LLEN) ? s[q] : 0.0f;
    }
    __syncthreads();

    int p0 = P0 + threadIdx.x * FIR_NP;
    float acc0 = 0.f, acc1 = 0.f, acc2 = 0.f, acc3 = 0.f;

    // prologue: q = p0+1 .. p0+3 (only outputs with p_i < q active)
    #pragma unroll
    for (int d = 1; d <= 3; ++d) {
        int q = p0 + d;
        int n = q / FRAME_SIZE;
        float sv = sbuf[q - (P0 + 1)];
        const float* fr = fb + (size_t)n * N_WINDOW + (N_WINDOW + p0 - q);
        if (d > 0) acc0 = fmaf(sv, fr[0], acc0);
        if (d > 1) acc1 = fmaf(sv, fr[1], acc1);
        if (d > 2) acc2 = fmaf(sv, fr[2], acc2);
    }

    // main: q = p0+4 .. min(p0+1024, L-1): all 4 outputs active, guard-free
    int qend = p0 + N_WINDOW;
    if (qend > LLEN - 1) qend = LLEN - 1;
    int q   = p0 + 4;
    int n   = q / FRAME_SIZE;
    int rem = q - n * FRAME_SIZE;
    for (; q <= qend; ++q) {
        float sv = sbuf[q - (P0 + 1)];
        const float* fr = fb + (size_t)n * N_WINDOW + (N_WINDOW + p0 - q);
        acc0 = fmaf(sv, fr[0], acc0);
        acc1 = fmaf(sv, fr[1], acc1);
        acc2 = fmaf(sv, fr[2], acc2);
        acc3 = fmaf(sv, fr[3], acc3);
        if (++rem == FRAME_SIZE) { rem = 0; ++n; }
    }

    // epilogue: q = p0+1025 .. p0+1027 (only outputs with q <= p_i+1024 active)
    #pragma unroll
    for (int d = 1; d <= 3; ++d) {
        int qq = p0 + N_WINDOW + d;
        if (qq <= LLEN - 1) {
            int n2 = qq / FRAME_SIZE;
            float sv = sbuf[qq - (P0 + 1)];
            const float* fr = fb + (size_t)n2 * N_WINDOW + (N_WINDOW + p0 - qq);
            if (d <= 1) acc1 = fmaf(sv, fr[1], acc1);
            if (d <= 2) acc2 = fmaf(sv, fr[2], acc2);
            acc3 = fmaf(sv, fr[3], acc3);
        }
    }

    float4 o;
    o.x = acc0; o.y = acc1; o.z = acc2; o.w = acc3;
    *(float4*)(out + (size_t)b * LLEN + p0) = o;
}

// ---------------- launcher ----------------
extern "C" void kernel_launch(void* const* d_in, const int* in_sizes, int n_in,
                              void* d_out, int out_size, void* d_ws, size_t ws_size,
                              hipStream_t stream) {
    const float* content  = (const float*)d_in[0];
    const float* f0       = (const float*)d_in[1];
    const float* energy   = (const float*)d_in[2];
    const float* spk      = (const float*)d_in[3];
    const float* source   = (const float*)d_in[4];
    const float* w_content= (const float*)d_in[5];
    const float* b_content= (const float*)d_in[6];
    const float* w_spk    = (const float*)d_in[7];
    const float* b_spk    = (const float*)d_in[8];
    const float* w_f0     = (const float*)d_in[9];
    const float* b_f0     = (const float*)d_in[10];
    const float* w_energy = (const float*)d_in[11];
    const float* b_energy = (const float*)d_in[12];
    const float* dw_w     = (const float*)d_in[13];
    const float* dw_b     = (const float*)d_in[14];
    const float* ln_g     = (const float*)d_in[15];
    const float* ln_b     = (const float*)d_in[16];
    const float* pw1_w    = (const float*)d_in[17];
    const float* pw1_b    = (const float*)d_in[18];
    const float* pw2_w    = (const float*)d_in[19];
    const float* pw2_b    = (const float*)d_in[20];
    const float* out_ln_g = (const float*)d_in[21];
    const float* out_ln_b = (const float*)d_in[22];
    const float* w_out    = (const float*)d_in[23];
    const float* b_out    = (const float*)d_in[24];
    float* out = (float*)d_out;

    char* base = (char*)d_ws;
    size_t off = 0;
    auto carve = [&](size_t bytes) -> char* {
        char* p = base + off;
        off = (off + bytes + 255) & ~(size_t)255;
        return p;
    };
    float*          xf     = (float*)         carve((size_t)BT * C_INT * 4);
    unsigned short* h_bf   = (unsigned short*)carve((size_t)BT * C_INT * 2);
    unsigned short* h1_bf  = (unsigned short*)carve((size_t)BT * D_MLP * 2);
    unsigned short* xnb    = (unsigned short*)carve((size_t)BT * C_INT * 2);
    float*          filt   = (float*)         carve((size_t)BT * N_WINDOW * 4);
    unsigned short* pw1b   = (unsigned short*)carve((size_t)N_LAYERS * D_MLP * C_INT * 2);
    unsigned short* pw2b   = (unsigned short*)carve((size_t)N_LAYERS * C_INT * D_MLP * 2);
    unsigned short* woutb  = (unsigned short*)carve((size_t)KSIZE * N_WINDOW * C_INT * 2);
    (void)ws_size; (void)in_sizes; (void)n_in; (void)out_size;

    // 1) weight conversion to bf16 (deterministic, every launch)
    {
        int n1 = N_LAYERS * D_MLP * C_INT;
        f2bf_copy_kernel<<<(n1 + 255) / 256, 256, 0, stream>>>(pw1_w, pw1b, n1);
        int n2 = N_LAYERS * C_INT * D_MLP;
        f2bf_copy_kernel<<<(n2 + 255) / 256, 256, 0, stream>>>(pw2_w, pw2b, n2);
        int n3 = N_WINDOW * C_INT * KSIZE;
        wout_tr_kernel<<<(n3 + 255) / 256, 256, 0, stream>>>(w_out, woutb);
    }

    // 2) conditioning trunk -> xf [BT][512]
    trunk_kernel<<<dim3(BT, C_INT / 256), 256, 0, stream>>>(
        content, f0, energy, spk, w_content, b_content, w_spk, b_spk,
        w_f0, b_f0, w_energy, b_energy, xf);

    // 3) ConvNeXt stack
    for (int l = 0; l < N_LAYERS; ++l) {
        dwln_kernel<<<BT, 256, 0, stream>>>(
            xf, dw_w + (size_t)l * C_INT * KSIZE, dw_b + (size_t)l * C_INT,
            ln_g + (size_t)l * C_INT, ln_b + (size_t)l * C_INT, h_bf);
        // pw1: [1536][512] x [4096][512]^T -> bf16 h1 [4096][1536] with GELU
        gemm_bf16_kernel<<<dim3(D_MLP / 128, BT / 64), 256, 0, stream>>>(
            pw1b + (size_t)l * D_MLP * C_INT, h_bf, pw1_b + (size_t)l * D_MLP,
            (void*)h1_bf, nullptr, D_MLP, C_INT, 0);
        // pw2: [512][1536] x [4096][1536]^T + residual -> f32 xf [4096][512]
        gemm_bf16_kernel<<<dim3(C_INT / 128, BT / 64), 256, 0, stream>>>(
            pw2b + (size_t)l * C_INT * D_MLP, h1_bf, pw2_b + (size_t)l * C_INT,
            (void*)xf, xf, C_INT, D_MLP, 1);
    }

    // 4) output LayerNorm -> bf16
    rowln_kernel<<<BT, 256, 0, stream>>>(xf, out_ln_g, out_ln_b, xnb);

    // 5) filter head (k=7 replicate-pad conv as 7 GEMMs) -> filt [4096][1024] f32
    gemm_filt_kernel<<<dim3(N_WINDOW / 128, BT / 64), 256, 0, stream>>>(
        woutb, xnb, b_out, filt);

    // 6) per-frame FIR + overlap-add -> out (B,1,L) f32
    fir_kernel<<<dim3(LLEN / FIR_PTILE, BATCH), FIR_BLOCK, 0, stream>>>(source, filt, out);
}
